// MultiheadAttention_32418413150804
// MI455X (gfx1250) — matrix operand order repaired and checked by reading, not compiled, not run on hardware
//
#include <hip/hip_runtime.h>
#include <stdint.h>

// Problem constants (match reference)
#define B_  32
#define TQ_ 1024
#define TK_ 1024
#define DK_ 64
#define LOG2E_ 1.4426950408889634f
// work in exp2 domain: z = s * log2(e)
#define KSCALE_ (0.125f * LOG2E_)                    // (1/sqrt(64)) * log2e
#define MASKZ_  (-4294967295.0f * LOG2E_)            // MASK_FILL * log2e

// LDS strides (elements of __bf16); rows padded to 16B multiples
#define KSTR  72   // 32 key rows x (64 d + 8 pad)   -> 144B rows
#define VTSTR 40   // 64 d rows  x (32 k + 8 pad)    -> 80B rows (V transposed)
#define PSTR  40   // per-wave P: 16 q rows x (32 k + 8 pad)

typedef __attribute__((ext_vector_type(16))) __bf16 v16bf;
typedef __attribute__((ext_vector_type(8)))  float  v8f;
typedef __attribute__((ext_vector_type(4)))  int    v4i;

#if defined(__HIP_DEVICE_COMPILE__) && \
    __has_builtin(__builtin_amdgcn_global_load_async_to_lds_b128) && \
    __has_builtin(__builtin_amdgcn_s_wait_asynccnt)
#define USE_ASYNC_MASK 1
typedef __attribute__((address_space(1))) v4i* gas_v4ip;   // global int4*
typedef __attribute__((address_space(3))) v4i* las_v4ip;   // LDS int4*
#else
#define USE_ASYNC_MASK 0
#endif

static __device__ __forceinline__ v8f vzero8() {
  v8f z;
#pragma unroll
  for (int i = 0; i < 8; ++i) z[i] = 0.0f;
  return z;
}

static __device__ __forceinline__ v8f wmma_bf16(v16bf a, v16bf b, v8f c) {
  // v_wmma_f32_16x16x32_bf16  (neg_a, A, neg_b, B, c_mod, C, reuse_a, reuse_b)
  return __builtin_amdgcn_wmma_f32_16x16x32_bf16(false, a, false, b, (short)0, c, false, false);
}

// Load 16 contiguous bf16 (32 bytes) from LDS into a fragment (2x ds_load_b128).
static __device__ __forceinline__ v16bf load32(const __bf16* p) {
  v16bf r;
#pragma unroll
  for (int i = 0; i < 16; ++i) r[i] = p[i];
  return r;
}

// A-fragment for Q (16x32 slice, k-step t): element i <-> d = 32t + 16*hi + i
static __device__ __forceinline__ v16bf qfrag(const float* __restrict__ qrow, int t, int hi) {
  const int d0 = 32 * t + 16 * hi;
  v16bf a;
#pragma unroll
  for (int i = 0; i < 8; ++i) {
    a[i]     = (__bf16)qrow[d0 + i];
    a[i + 8] = (__bf16)qrow[d0 + 8 + i];
  }
  return a;
}

// Cooperative stage of one 32-key chunk: K row-major bf16, V transposed bf16.
template <bool LOADV>
static __device__ __forceinline__ void stage_kv(const float* __restrict__ Kb,
                                                const float* __restrict__ Vb,
                                                int kc, int tid,
                                                __bf16* __restrict__ sK,
                                                __bf16* __restrict__ sVt) {
  const int row = tid >> 2;          // 0..31 key row within chunk
  const int cg  = (tid & 3) * 16;    // 0,16,32,48 column group
  {
    const float* kr = Kb + ((size_t)(kc + row)) * DK_ + cg;
    float t[16];
#pragma unroll
    for (int j = 0; j < 4; ++j) {
      float4 f = ((const float4*)kr)[j];
      t[4 * j + 0] = f.x; t[4 * j + 1] = f.y; t[4 * j + 2] = f.z; t[4 * j + 3] = f.w;
    }
    __bf16* dst = sK + row * KSTR + cg;
#pragma unroll
    for (int j = 0; j < 16; ++j) dst[j] = (__bf16)t[j];
    if (kc + 32 < TK_) __builtin_prefetch(kr + 32 * DK_, 0, 1);  // global_prefetch_b8
  }
  if (LOADV) {
    const float* vr = Vb + ((size_t)(kc + row)) * DK_ + cg;
    float t[16];
#pragma unroll
    for (int j = 0; j < 4; ++j) {
      float4 f = ((const float4*)vr)[j];
      t[4 * j + 0] = f.x; t[4 * j + 1] = f.y; t[4 * j + 2] = f.z; t[4 * j + 3] = f.w;
    }
#pragma unroll
    for (int j = 0; j < 16; ++j) sVt[(cg + j) * VTSTR + row] = (__bf16)t[j];
    if (kc + 32 < TK_) __builtin_prefetch(vr + 32 * DK_, 0, 1);
  }
}

__global__ __launch_bounds__(128) void
mha_fwd_kernel(const float* __restrict__ key, const float* __restrict__ value,
               const float* __restrict__ query, const unsigned char* __restrict__ mask,
               const float* __restrict__ qmask,
               float* __restrict__ outR, float* __restrict__ outA) {
  const int b    = blockIdx.y;
  const int tid  = threadIdx.x;
  const int wave = tid >> 5;
  const int lane = tid & 31;
  const int l15  = lane & 15;
  const int hi   = lane >> 4;
  const int q0   = blockIdx.x * 64 + wave * 16;

  __shared__ __align__(16) __bf16 sK[32 * KSTR];
  __shared__ __align__(16) __bf16 sVt[64 * VTSTR];
  __shared__ __align__(16) __bf16 sP[4][16 * PSTR];
  __shared__ __align__(16) unsigned char  sMask[64 * 32];        // 64 q rows x 32 keys / chunk
  __shared__ __align__(16) unsigned short sMbits[4 * 32 * 32];   // [wave][chunk][lane]

  const float*         Kb  = key   + (size_t)b * TK_ * DK_;
  const float*         Vb  = value + (size_t)b * TK_ * DK_;
  const float*       qrow  = query + ((size_t)b * TQ_ + q0 + l15) * DK_;
  const unsigned char* Mbw = mask  + ((size_t)b * TQ_ + q0) * (size_t)TK_;                 // this wave's 16 rows
  const unsigned char* Mbb = mask  + ((size_t)b * TQ_ + (size_t)blockIdx.x * 64) * TK_;    // block's 64 rows
  float*               Ab  = outA  + ((size_t)b * TQ_ + q0) * (size_t)TK_;

  // Q held in registers for the whole kernel (two A-fragments: d=0..31, d=32..63)
  const v16bf aq0 = qfrag(qrow, 0, hi);
  const v16bf aq1 = qfrag(qrow, 1, hi);

  // ---------------- Phase 1: per-lane online (max, sumexp) partials ----------------
  // Lane l15 only ever sees keys k ≡ (n0 + l15) mod 32; the 16 lanes of a half-group
  // partition the key axis, so no cross-lane reduction is needed until the very end.
  float pm[8], pl[8];
#pragma unroll
  for (int r = 0; r < 8; ++r) { pm[r] = -3.0e38f; pl[r] = 0.0f; }

  for (int kc = 0; kc < TK_; kc += 32) {
    __syncthreads();
    stage_kv<false>(Kb, Vb, kc, tid, sK, sVt);
#if USE_ASYNC_MASK
    {
      // Async DMA the block's 64x32 mask bytes for this chunk into LDS (ASYNCcnt path).
      const int mrw = tid >> 1, mof = (tid & 1) * 16;
      __builtin_amdgcn_global_load_async_to_lds_b128(
          (gas_v4ip)(Mbb + (size_t)mrw * TK_ + kc + mof),
          (las_v4ip)(sMask + mrw * 32 + mof), 0, 0);
      __builtin_amdgcn_s_wait_asynccnt(0);
    }
#endif
    __syncthreads();

    unsigned int bits = 0;  // 16 mask decisions for this lane/chunk
#pragma unroll
    for (int half = 0; half < 2; ++half) {
      const int n0 = half * 16;
      v8f c = vzero8();
      c = wmma_bf16(aq0, load32(&sK[(n0 + l15) * KSTR +      16 * hi]), c);
      c = wmma_bf16(aq1, load32(&sK[(n0 + l15) * KSTR + 32 + 16 * hi]), c);
#pragma unroll
      for (int r = 0; r < 8; ++r) {
        const int m = 8 * hi + r;
#if USE_ASYNC_MASK
        const unsigned char mb = sMask[(wave * 16 + m) * 32 + n0 + l15];
#else
        const unsigned char mb = Mbw[(size_t)m * TK_ + kc + n0 + l15];
#endif
        bits |= ((unsigned int)(mb != 0)) << (half * 8 + r);
        const float z  = mb ? MASKZ_ : c[r] * KSCALE_;
        const float nm = fmaxf(pm[r], z);
        pl[r] = pl[r] * exp2f(pm[r] - nm) + exp2f(z - nm);
        pm[r] = nm;
      }
    }
    sMbits[(wave * 32 + (kc >> 5)) * 32 + lane] = (unsigned short)bits;
  }

  // Single cross-lane merge of (max, sumexp) over the 16-lane half-group.
#pragma unroll
  for (int r = 0; r < 8; ++r) {
#pragma unroll
    for (int st = 1; st <= 8; st <<= 1) {
      const float om = __shfl_xor(pm[r], st, 32);
      const float ol = __shfl_xor(pl[r], st, 32);
      const float nm = fmaxf(pm[r], om);
      pl[r] = pl[r] * exp2f(pm[r] - nm) + ol * exp2f(om - nm);
      pm[r] = nm;
    }
  }

  // per-row factor: query_mask / row_sum  (folded into attn and O)
  float frow[8];
  {
    const float* qm = qmask + (size_t)b * TQ_ + q0;
#pragma unroll
    for (int r = 0; r < 8; ++r) frow[r] = qm[8 * hi + r] / pl[r];
  }

  // ---------------- Phase 2: recompute S, emit attn, accumulate O = P·V ----------------
  v8f o[4];
#pragma unroll
  for (int d = 0; d < 4; ++d) o[d] = vzero8();

  for (int kc = 0; kc < TK_; kc += 32) {
    __syncthreads();
    stage_kv<true>(Kb, Vb, kc, tid, sK, sVt);
    __syncthreads();
    const unsigned int bits = sMbits[(wave * 32 + (kc >> 5)) * 32 + lane];
#pragma unroll
    for (int half = 0; half < 2; ++half) {
      const int n0 = half * 16;
      v8f c = vzero8();
      c = wmma_bf16(aq0, load32(&sK[(n0 + l15) * KSTR +      16 * hi]), c);
      c = wmma_bf16(aq1, load32(&sK[(n0 + l15) * KSTR + 32 + 16 * hi]), c);
#pragma unroll
      for (int r = 0; r < 8; ++r) {
        const int m = 8 * hi + r;
        const float z = ((bits >> (half * 8 + r)) & 1u) ? MASKZ_ : c[r] * KSCALE_;
        const float p = exp2f(z - pm[r]) * frow[r];
        Ab[(size_t)m * TK_ + kc + n0 + l15] = p;                  // final attn, coalesced
        sP[wave][m * PSTR + n0 + l15] = (__bf16)p;                // stage for P·V A-fragment
      }
    }
    // A-fragment of P (wave-local LDS; compiler inserts s_wait_dscnt)
    v16bf ap;
    {
      const __bf16* pp = &sP[wave][l15 * PSTR + 16 * hi];
#pragma unroll
      for (int i = 0; i < 8; ++i) { ap[i] = pp[i]; ap[i + 8] = pp[8 + i]; }
    }
#pragma unroll
    for (int dt = 0; dt < 4; ++dt)
      o[dt] = wmma_bf16(ap, load32(&sVt[(dt * 16 + l15) * VTSTR + 16 * hi]), o[dt]);
  }

  // ---------------- Epilogue: write result (coalesced) ----------------
  float* Rb = outR + ((size_t)b * TQ_ + q0) * DK_;
#pragma unroll
  for (int dt = 0; dt < 4; ++dt)
#pragma unroll
    for (int r = 0; r < 8; ++r)
      Rb[(size_t)(8 * hi + r) * DK_ + dt * 16 + l15] = o[dt][r];
}

extern "C" void kernel_launch(void* const* d_in, const int* in_sizes, int n_in,
                              void* d_out, int out_size, void* d_ws, size_t ws_size,
                              hipStream_t stream) {
  (void)in_sizes; (void)n_in; (void)d_ws; (void)ws_size; (void)out_size;
  const float*         key   = (const float*)d_in[0];
  const float*         value = (const float*)d_in[1];
  const float*         query = (const float*)d_in[2];
  const unsigned char* mask  = (const unsigned char*)d_in[3];  // jnp.bool_ -> 1 byte/elem
  const float*         qmask = (const float*)d_in[4];

  float* outR = (float*)d_out;                                  // [B,TQ,DK]
  float* outA = outR + (size_t)B_ * TQ_ * DK_;                  // [B,TQ,TK]

  dim3 grid(TQ_ / 64, B_);   // 16 q-blocks x 32 batches
  dim3 block(128);           // 4 waves; each wave owns a 16-row q tile
  mha_fwd_kernel<<<grid, block, 0, stream>>>(key, value, query, mask, qmask, outR, outA);
}
